// MultiHeadedAttentionWithRelations_44006234915086
// MI455X (gfx1250) — matrix operand order repaired and checked by reading, not compiled, not run on hardware
//
#include <hip/hip_runtime.h>

// ---------------------------------------------------------------------------
// Relation-aware multi-head attention for MI455X (gfx1250, wave32, WMMA).
//
// Roofline: relation_k/relation_v = 1 GB fp32 must stream from HBM
// (~44us @ 23.3 TB/s). ~15 GFLOP total => ~330 TFLOP/s effective matrix rate
// needed to stay bandwidth bound => all matrix math on
// v_wmma_f32_16x16x32_bf16 (fp32 accumulate); fp32 -> bf16 conversion uses
// native bf16 casts (v_cvt_pk_bf16_f32) instead of integer rounding emulation.
// ---------------------------------------------------------------------------

constexpr int kB  = 8;
constexpr int kS  = 512;
constexpr int kD  = 512;   // d_model
constexpr int kH  = 8;
constexpr int kDK = 64;
constexpr int kQT = 16;    // queries per workgroup (one 16-row WMMA tile)
constexpr int kKT = 32;    // keys per streaming iteration (one WMMA K step)

typedef __attribute__((ext_vector_type(16))) __bf16         v16bf;
typedef __attribute__((ext_vector_type(4)))  __bf16         v4bf;
typedef __attribute__((ext_vector_type(16))) unsigned short v16u;
typedef __attribute__((ext_vector_type(8)))  unsigned short v8u;
typedef __attribute__((ext_vector_type(8)))  float          v8f;

#if defined(__AMDGCN__) && __has_builtin(__builtin_amdgcn_tensor_load_to_lds) && \
    __has_builtin(__builtin_amdgcn_s_wait_tensorcnt)
#define HAVE_TDM 1
typedef __attribute__((ext_vector_type(4))) unsigned int tdm_g0_t;
typedef __attribute__((ext_vector_type(8))) int          tdm_g1_t;
typedef __attribute__((ext_vector_type(4))) int          tdm_g23_t;
typedef __attribute__((ext_vector_type(8))) int          tdm_g4_t;
#else
#define HAVE_TDM 0
#endif

__device__ __forceinline__ unsigned short f2bf(float f) {
  __bf16 h = (__bf16)f;                       // native cvt (RNE)
  return __builtin_bit_cast(unsigned short, h);
}

__device__ __forceinline__ v8f wmma_bf16(v16bf a, v16bf b, v8f c) {
  // D = A(16x32 bf16) x B(32x16 bf16) + C(16x16 f32)
  return __builtin_amdgcn_wmma_f32_16x16x32_bf16(false, a, false, b,
                                                 (short)0, c, false, false);
}

// --- A fragment (16x32, M along lanes, K packed in regs) -------------------
// ISA layout: lanes 0-15: M=lane, K=0..7 in elems 0-7, K=16..23 in elems 8-15
//             lanes 16-31: M=lane-16, K=8..15 / 24..31.
__device__ __forceinline__ v16bf load_A_bf16(const unsigned short* base, int ldr, int lane) {
  int m = lane & 15, k0 = (lane >> 4) * 8;
  v8u lo = *(const v8u*)(base + (size_t)m * ldr + k0);
  v8u hi = *(const v8u*)(base + (size_t)m * ldr + 16 + k0);
  v16u r;
#pragma unroll
  for (int i = 0; i < 8; ++i) { r[i] = lo[i]; r[8 + i] = hi[i]; }
  return __builtin_bit_cast(v16bf, r);
}

__device__ __forceinline__ v16bf load_A_f32(const float* base, int ldr, int lane) {
  int m = lane & 15, k0 = (lane >> 4) * 8;
  const float4* p0 = (const float4*)(base + (size_t)m * ldr + k0);
  const float4* p1 = (const float4*)(base + (size_t)m * ldr + 16 + k0);
  float4 a0 = p0[0], a1 = p0[1], b0 = p1[0], b1 = p1[1];
  v16bf r;
  r[0]  = (__bf16)a0.x; r[1]  = (__bf16)a0.y; r[2]  = (__bf16)a0.z; r[3]  = (__bf16)a0.w;
  r[4]  = (__bf16)a1.x; r[5]  = (__bf16)a1.y; r[6]  = (__bf16)a1.z; r[7]  = (__bf16)a1.w;
  r[8]  = (__bf16)b0.x; r[9]  = (__bf16)b0.y; r[10] = (__bf16)b0.z; r[11] = (__bf16)b0.w;
  r[12] = (__bf16)b1.x; r[13] = (__bf16)b1.y; r[14] = (__bf16)b1.z; r[15] = (__bf16)b1.w;
  return r;
}

// --- B fragment (32x16) loaded from Bt row-major [N][K] --------------------
// ISA layout: lanes 0-15: N=lane, K=0..7 / 16..23; lanes 16-31: N=lane-16, K=8..15 / 24..31.
__device__ __forceinline__ v16bf load_Bt_bf16(const unsigned short* bt, int ldr, int lane) {
  int n = lane & 15, k0 = (lane >> 4) * 8;
  const v8u* p = (const v8u*)(bt + (size_t)n * ldr + k0);
  v8u lo = p[0], hi = p[2];
  v16u r;
#pragma unroll
  for (int i = 0; i < 8; ++i) { r[i] = lo[i]; r[8 + i] = hi[i]; }
  return __builtin_bit_cast(v16bf, r);
}

__device__ __forceinline__ v16bf load_Bt_f32(const float* bt, int ldr, int lane) {
  int n = lane & 15, k0 = (lane >> 4) * 8;
  const float4* p = (const float4*)(bt + (size_t)n * ldr + k0);
  float4 f0 = p[0], f1 = p[1], f2 = p[4], f3 = p[5];
  v16bf r;
  r[0]  = (__bf16)f0.x; r[1]  = (__bf16)f0.y; r[2]  = (__bf16)f0.z; r[3]  = (__bf16)f0.w;
  r[4]  = (__bf16)f1.x; r[5]  = (__bf16)f1.y; r[6]  = (__bf16)f1.z; r[7]  = (__bf16)f1.w;
  r[8]  = (__bf16)f2.x; r[9]  = (__bf16)f2.y; r[10] = (__bf16)f2.z; r[11] = (__bf16)f2.w;
  r[12] = (__bf16)f3.x; r[13] = (__bf16)f3.y; r[14] = (__bf16)f3.z; r[15] = (__bf16)f3.w;
  return r;
}

// ---------------------------------------------------------------------------
// Kernel 1: Y = X @ W^T + b  (X:[4096,512] f32, W:[512,512] f32) -> bf16.
// vmode 0: store [b,h,s,dk] (Q,K);  vmode 1: store transposed [b,h,dk,s] (V).
// ---------------------------------------------------------------------------
__global__ void __launch_bounds__(256)
proj_gemm_kernel(const float* __restrict__ X, const float* __restrict__ W,
                 const float* __restrict__ bias, unsigned short* __restrict__ out,
                 int vmode) {
  int lane = threadIdx.x & 31, wv = threadIdx.x >> 5;
  int tile = blockIdx.x * 8 + wv;          // 8192 tiles = 256(M) x 32(N)
  int tM = tile >> 5, tN = tile & 31;
  const float* Xb = X + (size_t)tM * 16 * kD;
  const float* Wb = W + (size_t)tN * 16 * kD;   // Bt[n][k] = W[n][k]
  v8f acc;
#pragma unroll
  for (int i = 0; i < 8; ++i) acc[i] = 0.f;
#pragma unroll 4
  for (int k = 0; k < kD; k += 32)
    acc = wmma_bf16(load_A_f32(Xb + k, kD, lane), load_Bt_f32(Wb + k, kD, lane), acc);

  int n = tN * 16 + (lane & 15);
  int jb = (lane >> 4) * 8;
  float bs = bias[n];
  int h = n >> 6, dk = n & 63;
#pragma unroll
  for (int j = 0; j < 8; ++j) {
    int m = tM * 16 + jb + j;
    int bb = m >> 9, s = m & 511;
    unsigned short y = f2bf(acc[j] + bs);
    if (vmode) out[(((size_t)bb * kH + h) * kDK + dk) * kS + s] = y;
    else       out[(((size_t)bb * kH + h) * kS + s) * kDK + dk] = y;
  }
}

// ---------------------------------------------------------------------------
// Kernel 2: fused relation-aware attention with online softmax.
// Grid: (S/16 q-tiles, B). Block: 256 threads = 8 waves.
// Wave roles per iteration: "q-role" (2 queries each: relK scores + relV
// staging + P@relV) and "head-role" (1 head each: QK^T + softmax + P@V).
// ---------------------------------------------------------------------------
__global__ void __launch_bounds__(256)
rel_attn_kernel(const unsigned short* __restrict__ Qh,   // [b,h,s,dk] bf16
                const unsigned short* __restrict__ Kh,   // [b,h,s,dk] bf16
                const unsigned short* __restrict__ Vt,   // [b,h,dk,s] bf16
                const float* __restrict__ relK,          // [b,q,k,dk] f32
                const float* __restrict__ relV,          // [b,q,k,dk] f32
                unsigned short* __restrict__ Xout) {     // [b,s,d] bf16
  extern __shared__ char smem[];
  unsigned short* lds_qall = (unsigned short*)smem;            // [8h][16q][64d]
  unsigned short* lds_rvt  = lds_qall + kH * kQT * kDK;        // [16q][64d][32k]
  unsigned short* lds_p    = lds_rvt + kQT * kDK * kKT;        // [8h][16q][32k]
  float* lds_rels  = (float*)(lds_p + kH * kQT * kKT);         // [16q][8h][32k]
  float* lds_alpha = lds_rels + kQT * kH * kKT;                // [8h][16q]
  float* lds_l     = lds_alpha + kH * kQT;                     // [8h][16q]
  float* lds_x     = lds_l + kH * kQT;                         // [16q][8h][64d]

  int tid = threadIdx.x, lane = tid & 31, wv = tid >> 5;
  int b = blockIdx.y, qbase = blockIdx.x * kQT;
  int h = wv;                       // head-role
  int jb  = (lane >> 4) * 8;        // C-fragment row base
  int col = lane & 15;              // C-fragment column
  int laneH = (lane & 16) | (lane & 7);  // lane remap: clamp A rows to 8 heads

  // ---- Stage Q (all heads) for this q-tile into LDS as [h][q][d]. ---------
  // Per-head source block (16 rows x 64 dk) is 1024 contiguous bf16, 8 rows
  // stride S*DK apart: an ideal 2-D TDM tile.
#if HAVE_TDM
  if (wv == 0) {
    unsigned long long ga =
        (unsigned long long)(const char*)(Qh + (((size_t)b * kH) * kS + qbase) * kDK);
    unsigned lds_off =
        (unsigned)(unsigned long long)(__attribute__((address_space(3))) char*)(void*)lds_qall;
    tdm_g0_t g0;
    g0[0] = 1u;                                   // count=1, user mode
    g0[1] = lds_off;                              // lds_addr (bytes)
    g0[2] = (unsigned)(ga & 0xFFFFFFFFull);       // global_addr[31:0]
    g0[3] = (unsigned)((ga >> 32) & 0x01FFFFFFull) | 0x80000000u; // addr[56:32] | type=2
    tdm_g1_t g1;
    g1[0] = (int)(1u << 16);                      // data_size=1 (2 bytes), no mask
    g1[1] = (int)0x80000000u;                     // tensor_dim0 = 0x8000 (=S*DK elems)
    g1[2] = (int)(8u << 16);                      // tensor_dim1 = 8 rows (heads)
    g1[3] = (int)(1024u << 16);                   // tile_dim0 = 1024 elems (16q*64d)
    g1[4] = 8;                                    // tile_dim1 = 8
    g1[5] = (int)(kS * kDK);                      // tensor_dim0_stride = 32768 elems
    g1[6] = 0;
    g1[7] = 0;
    tdm_g23_t gz; gz[0] = 0; gz[1] = 0; gz[2] = 0; gz[3] = 0;
    tdm_g4_t  gz8;
#pragma unroll
    for (int i = 0; i < 8; ++i) gz8[i] = 0;
    // 6-arg toolchain form: (g0, g1, g2, g3, g4, cpol)
    __builtin_amdgcn_tensor_load_to_lds(g0, g1, gz, gz, gz8, 0);
    __builtin_amdgcn_s_wait_tensorcnt(0);
  }
#else
  for (int e = tid; e < kH * kQT * kDK; e += 256) {
    int d = e & 63, q = (e >> 6) & 15, hh = e >> 10;
    lds_qall[e] = Qh[(((size_t)b * kH + hh) * kS + qbase + q) * kDK + d];
  }
#endif
  __syncthreads();

  // Loop-invariant A fragments.
  const unsigned short* Qhb = Qh + (((size_t)b * kH + h) * kS + qbase) * kDK;
  v16bf aQ0 = load_A_bf16(Qhb,      kDK, lane);     // d = 0..31
  v16bf aQ1 = load_A_bf16(Qhb + 32, kDK, lane);     // d = 32..63
  v16bf aHq[2][2];   // rows = heads, for q0,q1: element lds_qall[(h*16+q)*64+d]
#pragma unroll
  for (int qq = 0; qq < 2; ++qq)
#pragma unroll
    for (int ks = 0; ks < 2; ++ks)
      aHq[qq][ks] = load_A_bf16(lds_qall + (2 * wv + qq) * kDK + ks * 32,
                                kQT * kDK, laneH);

  v8f zero;
#pragma unroll
  for (int i = 0; i < 8; ++i) zero[i] = 0.f;
  float m_run[8], l_run[8];
  v8f accv[4], accr0[4], accr1[4];
#pragma unroll
  for (int t = 0; t < 4; ++t) { accv[t] = zero; accr0[t] = zero; accr1[t] = zero; }
#pragma unroll
  for (int j = 0; j < 8; ++j) { m_run[j] = -1e30f; l_run[j] = 0.f; }

  for (int kt = 0; kt < kS / kKT; ++kt) {
    int kb = kt * kKT;

    // ---- Phase 1 (q-role): stage relV^T tile + relation scores ------------
#pragma unroll
    for (int qq = 0; qq < 2; ++qq) {
      int q = 2 * wv + qq, qg = qbase + q;
      const float* rVq = relV + (((size_t)b * kS + qg) * kS + kb) * kDK;
      const float* rKq = relK + (((size_t)b * kS + qg) * kS + kb) * kDK;
      if (kt + 1 < kS / kKT) {                     // hide next tile's latency
        __builtin_prefetch(rVq + kKT * kDK, 0, 1); // -> global_prefetch_b8
        __builtin_prefetch(rKq + kKT * kDK, 0, 1);
      }
      // relV[32k][64d] f32 -> LDS relV^T [64d][32k] bf16, packed ds_store_b64
      {
        int d4 = (lane & 15) * 4;
        int kk0 = (lane >> 4) * 4;
        for (int kk = kk0; kk < kKT; kk += 8) {
          float4 fa = *(const float4*)(rVq + (kk + 0) * kDK + d4);
          float4 fb = *(const float4*)(rVq + (kk + 1) * kDK + d4);
          float4 fc = *(const float4*)(rVq + (kk + 2) * kDK + d4);
          float4 fd = *(const float4*)(rVq + (kk + 3) * kDK + d4);
          unsigned short* dst = lds_rvt + ((size_t)q * kDK + d4) * kKT + kk;
          *(v4bf*)(dst + 0 * kKT) = (v4bf){(__bf16)fa.x, (__bf16)fb.x, (__bf16)fc.x, (__bf16)fd.x};
          *(v4bf*)(dst + 1 * kKT) = (v4bf){(__bf16)fa.y, (__bf16)fb.y, (__bf16)fc.y, (__bf16)fd.y};
          *(v4bf*)(dst + 2 * kKT) = (v4bf){(__bf16)fa.z, (__bf16)fb.z, (__bf16)fc.z, (__bf16)fd.z};
          *(v4bf*)(dst + 3 * kKT) = (v4bf){(__bf16)fa.w, (__bf16)fb.w, (__bf16)fc.w, (__bf16)fd.w};
        }
      }
      // rel score [8h x 32k] for this q: A rows = heads, Bt rows = keys.
#pragma unroll
      for (int nt = 0; nt < 2; ++nt) {
        v8f c = zero;
        c = wmma_bf16(aHq[qq][0], load_Bt_f32(rKq + nt * 16 * kDK,      kDK, lane), c);
        c = wmma_bf16(aHq[qq][1], load_Bt_f32(rKq + nt * 16 * kDK + 32, kDK, lane), c);
        if (lane < 16) {              // rows 0..7 = valid heads
#pragma unroll
          for (int j = 0; j < 8; ++j)
            lds_rels[((size_t)q * kH + j) * kKT + nt * 16 + col] = c[j];
        }
      }
    }
    __syncthreads();

    // ---- Phase 2 (head-role): QK^T + rel + online softmax -----------------
    const unsigned short* Kq = Kh + (((size_t)b * kH + h) * kS + kb) * kDK;
    v8f s0 = zero, s1 = zero;
    s0 = wmma_bf16(aQ0, load_Bt_bf16(Kq,                 kDK, lane), s0);
    s0 = wmma_bf16(aQ1, load_Bt_bf16(Kq + 32,            kDK, lane), s0);
    s1 = wmma_bf16(aQ0, load_Bt_bf16(Kq + 16 * kDK,      kDK, lane), s1);
    s1 = wmma_bf16(aQ1, load_Bt_bf16(Kq + 16 * kDK + 32, kDK, lane), s1);
#pragma unroll
    for (int j = 0; j < 8; ++j) {
      int q = jb + j;
      float sc0 = (s0[j] + lds_rels[((size_t)q * kH + h) * kKT + col]) * 0.125f;
      float sc1 = (s1[j] + lds_rels[((size_t)q * kH + h) * kKT + 16 + col]) * 0.125f;
      float rm = fmaxf(sc0, sc1);                  // row-max across 16 lanes
      rm = fmaxf(rm, __shfl_xor(rm, 1, 32));
      rm = fmaxf(rm, __shfl_xor(rm, 2, 32));
      rm = fmaxf(rm, __shfl_xor(rm, 4, 32));
      rm = fmaxf(rm, __shfl_xor(rm, 8, 32));
      float mnew = fmaxf(m_run[j], rm);
      float al = __expf(m_run[j] - mnew);
      m_run[j] = mnew;
      float p0 = __expf(sc0 - mnew), p1 = __expf(sc1 - mnew);
      float rs = p0 + p1;
      rs += __shfl_xor(rs, 1, 32);
      rs += __shfl_xor(rs, 2, 32);
      rs += __shfl_xor(rs, 4, 32);
      rs += __shfl_xor(rs, 8, 32);
      l_run[j] = l_run[j] * al + rs;
#pragma unroll
      for (int t = 0; t < 4; ++t) accv[t][j] *= al;   // rescale running PV
      lds_p[((size_t)h * kQT + q) * kKT + col]      = f2bf(p0);
      lds_p[((size_t)h * kQT + q) * kKT + 16 + col] = f2bf(p1);
      if (col == 0) lds_alpha[h * kQT + q] = al;
    }
    __syncthreads();

    // ---- Phase 3a (head-role): PV with pre-transposed V -------------------
    v16bf aP = load_A_bf16(lds_p + (size_t)h * kQT * kKT, kKT, lane); // rows=q
    const unsigned short* Vb = Vt + (((size_t)b * kH + h) * kDK) * kS + kb;
#pragma unroll
    for (int t = 0; t < 4; ++t)
      accv[t] = wmma_bf16(aP, load_Bt_bf16(Vb + (size_t)t * 16 * kS, kS, lane), accv[t]);

    // ---- Phase 3b (q-role): P(all heads) @ relV ---------------------------
#pragma unroll
    for (int qq = 0; qq < 2; ++qq) {
      int q = 2 * wv + qq;
      v8f* accr = qq ? accr1 : accr0;
#pragma unroll
      for (int j = 0; j < 8; ++j) {
        float al = lds_alpha[((jb + j) & 7) * kQT + q];
#pragma unroll
        for (int t = 0; t < 4; ++t) accr[t][j] *= al;
      }
      // A rows = heads: element lds_p[(hrow*16 + q)*32 + k], row stride 16*32.
      v16bf aPH = load_A_bf16(lds_p + (size_t)q * kKT, kQT * kKT, laneH);
#pragma unroll
      for (int t = 0; t < 4; ++t)
        accr[t] = wmma_bf16(
            aPH, load_Bt_bf16(lds_rvt + ((size_t)q * kDK + t * 16) * kKT, kKT, lane),
            accr[t]);
    }
    __syncthreads();   // protect lds_p / lds_rvt / lds_rels for next tile
  }

  // ---- Epilogue: combine main + relation outputs through LDS --------------
#pragma unroll
  for (int j = 0; j < 8; ++j) {
    int q = jb + j;
    float inv = 1.f / l_run[j];
#pragma unroll
    for (int t = 0; t < 4; ++t)
      lds_x[((size_t)q * kH + h) * kDK + t * 16 + col] = accv[t][j] * inv;
    if (col == 0) lds_l[h * kQT + q] = l_run[j];
  }
  __syncthreads();
  if (lane < 16) {                     // accr rows 0..7 = valid heads
#pragma unroll
    for (int qq = 0; qq < 2; ++qq) {
      int q = 2 * wv + qq;
      v8f* accr = qq ? accr1 : accr0;
#pragma unroll
      for (int j = 0; j < 8; ++j) {
        float inv = 1.f / lds_l[j * kQT + q];
#pragma unroll
        for (int t = 0; t < 4; ++t)
          lds_x[((size_t)q * kH + j) * kDK + t * 16 + col] += accr[t][j] * inv;
      }
    }
  }
  __syncthreads();
  for (int e = tid; e < kQT * kD; e += 256) {
    int q = e >> 9, c = e & 511;       // [q][h][d] flat == q*512 + (h*64+d)
    Xout[((size_t)b * kS + qbase + q) * kD + c] = f2bf(lds_x[e]);
  }
}

// ---------------------------------------------------------------------------
// Kernel 3: final projection  out = X @ Wo^T + bo  (bf16 x f32->bf16, f32 out)
// ---------------------------------------------------------------------------
__global__ void __launch_bounds__(256)
out_proj_kernel(const unsigned short* __restrict__ Xb16, const float* __restrict__ Wo,
                const float* __restrict__ bo, float* __restrict__ out) {
  int lane = threadIdx.x & 31, wv = threadIdx.x >> 5;
  int tile = blockIdx.x * 8 + wv;
  int tM = tile >> 5, tN = tile & 31;
  const unsigned short* Xb = Xb16 + (size_t)tM * 16 * kD;
  const float* Wb = Wo + (size_t)tN * 16 * kD;
  v8f acc;
#pragma unroll
  for (int i = 0; i < 8; ++i) acc[i] = 0.f;
#pragma unroll 4
  for (int k = 0; k < kD; k += 32)
    acc = wmma_bf16(load_A_bf16(Xb + k, kD, lane), load_Bt_f32(Wb + k, kD, lane), acc);
  int n = tN * 16 + (lane & 15);
  int jb = (lane >> 4) * 8;
  float bs = bo[n];
#pragma unroll
  for (int j = 0; j < 8; ++j) {
    int m = tM * 16 + jb + j;
    out[(size_t)m * kD + n] = acc[j] + bs;
  }
}

// ---------------------------------------------------------------------------
extern "C" void kernel_launch(void* const* d_in, const int* in_sizes, int n_in,
                              void* d_out, int out_size, void* d_ws, size_t ws_size,
                              hipStream_t stream) {
  (void)in_sizes; (void)n_in; (void)out_size; (void)ws_size;
  const float* query = (const float*)d_in[0];
  const float* key_  = (const float*)d_in[1];
  const float* value = (const float*)d_in[2];
  const float* relK  = (const float*)d_in[3];
  const float* relV  = (const float*)d_in[4];
  const float* Wq = (const float*)d_in[6];
  const float* bq = (const float*)d_in[7];
  const float* Wk = (const float*)d_in[8];
  const float* bk = (const float*)d_in[9];
  const float* Wv = (const float*)d_in[10];
  const float* bv = (const float*)d_in[11];
  const float* Wo = (const float*)d_in[12];
  const float* bo = (const float*)d_in[13];

  // Workspace: Qh, Kh, Vt (V transposed), Xout — all bf16, 4 MB each = 16 MB.
  unsigned short* Qh = (unsigned short*)d_ws;
  unsigned short* Kh = Qh + (size_t)kB * kS * kD;
  unsigned short* Vt = Kh + (size_t)kB * kS * kD;
  unsigned short* Xo = Vt + (size_t)kB * kS * kD;

  dim3 blk(256);
  proj_gemm_kernel<<<1024, blk, 0, stream>>>(query, Wq, bq, Qh, 0);
  proj_gemm_kernel<<<1024, blk, 0, stream>>>(key_,  Wk, bk, Kh, 0);
  proj_gemm_kernel<<<1024, blk, 0, stream>>>(value, Wv, bv, Vt, 1);

  size_t lds = (size_t)(kH * kQT * kDK + kQT * kDK * kKT + kH * kQT * kKT) * sizeof(unsigned short)
             + (size_t)(kQT * kH * kKT + 2 * kH * kQT + kQT * kH * kDK) * sizeof(float);
  rel_attn_kernel<<<dim3(kS / kQT, kB), blk, lds, stream>>>(Qh, Kh, Vt, relK, relV, Xo);

  out_proj_kernel<<<1024, blk, 0, stream>>>(Xo, Wo, bo, (float*)d_out);
}
